// NMFInformedVQ_87187836109018
// MI455X (gfx1250) — matrix operand order repaired and checked by reading, not compiled, not run on hardware
//
#include <hip/hip_runtime.h>
#include <hip/hip_bf16.h>

typedef __attribute__((ext_vector_type(16))) _Float16 v16h;
typedef __attribute__((ext_vector_type(8)))  _Float16 v8h;
typedef __attribute__((ext_vector_type(8)))  float    v8f;

#define VQ_D     200
#define VQ_DPAD  224           // K padded to multiple of 32 for 16x16x32 WMMA
#define VQ_KT    (VQ_DPAD/32)  // 7 k-tiles
#define VQ_CHUNK 4             // code tiles per accumulation chunk
#define VQ_MT    2             // row tiles per wave (M = 32)
#define VQ_WAVES 4             // waves per block

// ---------------------------------------------------------------------------
// Kernel 1: L2-normalize active codebook rows -> f16, padded [rowsAlloc][224].
// Rows >= active are zeroed (so padded WMMA tiles contribute sim = 0).
// Also zeroes the loss accumulator. One wave per codebook row.
// ---------------------------------------------------------------------------
__global__ void vq_norm_codes_87187836109018(const float* __restrict__ W,
                                             const int* __restrict__ activeP,
                                             _Float16* __restrict__ wn,
                                             float* __restrict__ lossPtr) {
    const int row  = blockIdx.x;
    const int lane = threadIdx.x;
    if (row == 0 && lane == 0) *lossPtr = 0.0f;
    const int A = *activeP;
    if (row >= A) {
        for (int k = lane; k < VQ_DPAD; k += 32)
            wn[(size_t)row * VQ_DPAD + k] = (_Float16)0.0f;
        return;
    }
    float s = 0.0f;
    for (int k = lane; k < VQ_D; k += 32) {
        float v = W[(size_t)row * VQ_D + k];
        s += v * v;
    }
    for (int off = 16; off; off >>= 1) s += __shfl_xor(s, off, 32);
    const float inv = 1.0f / fmaxf(sqrtf(s), 1e-12f);
    for (int k = lane; k < VQ_DPAD; k += 32) {
        float v = (k < VQ_D) ? W[(size_t)row * VQ_D + k] * inv : 0.0f;
        wn[(size_t)row * VQ_DPAD + k] = (_Float16)v;
    }
}

// ---------------------------------------------------------------------------
// Kernel 2: one wave per 32 rows (two 16-row WMMA tiles). Normalize rows into
// LDS f16, dense WMMA loop vs zero-padded codebook (batched B loads per
// k-step), running argmax on the f32 accumulators, then exact-f32 gather+loss.
// ---------------------------------------------------------------------------
__global__ void __launch_bounds__(32 * VQ_WAVES)
vq_main_87187836109018(const float* __restrict__ X,
                       const float* __restrict__ W,
                       const int* __restrict__ activeP,
                       const _Float16* __restrict__ wn,
                       float* __restrict__ out,
                       float* __restrict__ lossPtr,
                       float* __restrict__ idxOut,
                       long N, float lossScale) {
    __shared__ __align__(16) _Float16 lds_a[VQ_WAVES][32][VQ_DPAD]; // 57344 B
    __shared__ int lds_idx[VQ_WAVES][32];

    const int  lane    = threadIdx.x & 31;
    const int  wv      = threadIdx.x >> 5;
    const long rowBase = ((long)blockIdx.x * VQ_WAVES + wv) * 32;
    if (rowBase >= N) return;                  // wave-uniform

    const int A        = *activeP;
    const int tilesA   = (A + 15) >> 4;
    const int tilesPad = (tilesA + VQ_CHUNK - 1) & ~(VQ_CHUNK - 1);

    // ---- normalize: lane L owns row rowBase+L; write f16 row into LDS ----
    {
        const bool valid = (rowBase + lane) < N;
        _Float16* dst = &lds_a[wv][lane][0];
        if (valid) {
            const float4* xv = (const float4*)(X + (rowBase + lane) * VQ_D);
            float s = 0.0f;
            for (int k = 0; k < VQ_D / 4; ++k) {
                float4 v = xv[k];
                s += v.x * v.x + v.y * v.y + v.z * v.z + v.w * v.w;
            }
            const float inv = 1.0f / fmaxf(sqrtf(s), 1e-12f);
            const float* xr = X + (rowBase + lane) * VQ_D;
            for (int k = 0; k < VQ_D; ++k) dst[k] = (_Float16)(xr[k] * inv);
            for (int k = VQ_D; k < VQ_DPAD; ++k) dst[k] = (_Float16)0.0f;
        } else {
            for (int k = 0; k < VQ_DPAD; ++k) dst[k] = (_Float16)0.0f;
        }
    }
    // same-wave LDS producer/consumer: per-wave LDS ops are in order.

    // ---- dense WMMA GEMM + running argmax ----
    float bestV[VQ_MT][8];
    int   bestC[VQ_MT][8];
#pragma unroll
    for (int m = 0; m < VQ_MT; ++m)
#pragma unroll
        for (int i = 0; i < 8; ++i) { bestV[m][i] = -3.4e38f; bestC[m][i] = 0; }

    const int r    = lane & 15;
    const int half = lane >> 4;
    const int aKb  = half ? 16 : 0;  // A-frag K base within 32-chunk (contiguous, matches B)
    const int bKb  = half ? 16 : 0;  // B-frag K base within 32-chunk (contiguous, matches A)

    for (int ct0 = 0; ct0 < tilesPad; ct0 += VQ_CHUNK) {
        v8f acc[VQ_MT][VQ_CHUNK] = {};

        // Base pointer per code tile of this chunk; kt advances via immediates.
        const _Float16* bp[VQ_CHUNK];
#pragma unroll
        for (int t = 0; t < VQ_CHUNK; ++t)
            bp[t] = wn + (size_t)((ct0 + t) * 16 + r) * VQ_DPAD + bKb;

        for (int kt = 0; kt < VQ_KT; ++kt) {
            const int k0 = kt * 32;
            v16h a[VQ_MT];
#pragma unroll
            for (int m = 0; m < VQ_MT; ++m) {
                v8h lo = *(const v8h*)&lds_a[wv][m * 16 + r][k0 + aKb];
                v8h hi = *(const v8h*)&lds_a[wv][m * 16 + r][k0 + 8 + aKb];
                a[m] = __builtin_shufflevector(lo, hi,
                          0,1,2,3,4,5,6,7,8,9,10,11,12,13,14,15);
            }
            // Batch all B loads of this k-step (8 x b128 in one clause), then
            // run the 8-WMMA burst back-to-back behind a single wait.
            v16h b[VQ_CHUNK];
#pragma unroll
            for (int t = 0; t < VQ_CHUNK; ++t)
                b[t] = *(const v16h*)(bp[t] + k0);
#pragma unroll
            for (int t = 0; t < VQ_CHUNK; ++t)
#pragma unroll
                for (int m = 0; m < VQ_MT; ++m)
                    acc[m][t] = __builtin_amdgcn_wmma_f32_16x16x32_f16(
                        false, a[m], false, b[t], (short)0, acc[m][t], false, false);
        }
#pragma unroll
        for (int t = 0; t < VQ_CHUNK; ++t) {
            const int c = ((ct0 + t) << 4) + r;
            if (c < A) {                      // per-lane mask, outside WMMA
#pragma unroll
                for (int m = 0; m < VQ_MT; ++m)
#pragma unroll
                    for (int i = 0; i < 8; ++i) {
                        float v = acc[m][t][i];
                        if (v > bestV[m][i]) { bestV[m][i] = v; bestC[m][i] = c; }
                    }
            }
        }
    }

    // ---- reduce argmax across the 16 lanes sharing each row half ----
#pragma unroll
    for (int m = 0; m < VQ_MT; ++m) {
#pragma unroll
        for (int i = 0; i < 8; ++i) {
            float bv = bestV[m][i];
            int   bc = bestC[m][i];
            for (int off = 8; off; off >>= 1) {
                float ov = __shfl_xor(bv, off, 32);
                int   oc = __shfl_xor(bc, off, 32);
                if (ov > bv || (ov == bv && oc < bc)) { bv = ov; bc = oc; }
            }
            bestV[m][i] = bv; bestC[m][i] = bc;
        }
    }
    if ((lane & 15) == 0) {
#pragma unroll
        for (int m = 0; m < VQ_MT; ++m)
#pragma unroll
            for (int i = 0; i < 8; ++i)
                lds_idx[wv][m * 16 + half * 8 + i] = bestC[m][i];
    }

    // ---- gather W[idx] (exact f32), write output + indices, loss ----
    float lsum = 0.0f;
    {
        const long row   = rowBase + lane;
        const bool valid = row < N;
        if (valid) {
            const int idx = lds_idx[wv][lane];          // same-wave LDS ordering
            const float4* qv = (const float4*)(W + (size_t)idx * VQ_D);
            const float4* xv = (const float4*)(X + row * VQ_D);
            float4*       ov = (float4*)(out + row * VQ_D);
            for (int k = 0; k < VQ_D / 4; ++k) {
                float4 q = qv[k];
                float4 x = xv[k];
                ov[k] = q;
                float dx = q.x - x.x, dy = q.y - x.y, dz = q.z - x.z, dw = q.w - x.w;
                lsum += dx * dx + dy * dy + dz * dz + dw * dw;
            }
            idxOut[row] = (float)idx;
        }
    }
    for (int off = 16; off; off >>= 1) lsum += __shfl_xor(lsum, off, 32);
    if (lane == 0) atomicAdd(lossPtr, lsum * lossScale);
}

// ---------------------------------------------------------------------------
extern "C" void kernel_launch(void* const* d_in, const int* in_sizes, int n_in,
                              void* d_out, int out_size, void* d_ws, size_t ws_size,
                              hipStream_t stream) {
    const float* X       = (const float*)d_in[0];
    const float* W       = (const float*)d_in[1];
    const int*   activeP = (const int*)d_in[2];

    const long N        = (long)in_sizes[0] / VQ_D;   // 262144 rows
    const int  numCodes = in_sizes[1] / VQ_D;         // 1024

    // Allocate/zero enough padded codebook rows that rounding the active tile
    // count up to a multiple of VQ_CHUNK always stays inside the buffer.
    const int tilesCodes = (numCodes + 15) >> 4;
    const int tilesAlloc = (tilesCodes + VQ_CHUNK - 1) & ~(VQ_CHUNK - 1);
    const int rowsAlloc  = tilesAlloc << 4;           // 1024 for this problem

    _Float16* wn = (_Float16*)d_ws;                   // [rowsAlloc][224] f16

    float* out     = (float*)d_out;
    float* lossPtr = out + (size_t)N * VQ_D;
    float* idxOut  = lossPtr + 1;

    const float lossScale = 0.25f / (float)((double)N * (double)VQ_D);

    vq_norm_codes_87187836109018<<<rowsAlloc, 32, 0, stream>>>(W, activeP, wn, lossPtr);

    const long waveTiles = (N + 31) / 32;             // 32 rows per wave
    const long blocks    = (waveTiles + VQ_WAVES - 1) / VQ_WAVES;
    vq_main_87187836109018<<<(int)blocks, 32 * VQ_WAVES, 0, stream>>>(
        X, W, activeP, wn, out, lossPtr, idxOut, N, lossScale);
}
